// AttentionOnlyModel_87050397155509
// MI455X (gfx1250) — matrix operand order repaired and checked by reading, not compiled, not run on hardware
//
#include <hip/hip_runtime.h>

typedef __attribute__((ext_vector_type(16))) _Float16 v16h;
typedef __attribute__((ext_vector_type(8)))  float    v8f;

namespace {
constexpr int   kB          = 8;
constexpr int   kN          = 32;
constexpr int   kH          = 128;
constexpr int   kKV         = 4096;
constexpr int   kWaves      = 8;
constexpr int   kRowsPerWave = kKV / kWaves;   // 512
constexpr int   kTile       = 32;              // KV rows per iteration
constexpr float kScale      = 0.08838834764831845f;  // 128^-0.5
}

__global__ __launch_bounds__(kWaves * 32)
void fa_decode_wmma(const float* __restrict__ qg,
                    const float* __restrict__ kg,
                    const float* __restrict__ vg,
                    const float* __restrict__ mg,
                    float* __restrict__ og)
{
  const int bn   = blockIdx.x;           // head index = b*kN + n
  const int b    = bn >> 5;              // kN == 32
  const int tid  = threadIdx.x;
  const int lane = tid & 31;
  const int wave = tid >> 5;
  const int half = lane >> 4;            // 0: lanes 0-15, 1: lanes 16-31
  const int l16  = lane & 15;

  const float* qp = qg + (size_t)bn * kH;
  const float* kp = kg + (size_t)bn * kKV * kH;
  const float* vp = vg + (size_t)bn * kKV * kH;
  const float* mp = mg + (size_t)b * kKV;

  // ---- A fragments of scaled q, one per 32-wide H chunk --------------------
  // A 16x32, contiguous k order used here: lanes 0-15, elem e -> K=e;
  // lanes 16-31 add 16. All rows identical (q broadcast).
  v16h aq[4];
#pragma unroll
  for (int c = 0; c < 4; ++c) {
#pragma unroll
    for (int e = 0; e < 16; ++e) {
      const int h = c * 32 + half * 16 + e;
      aq[c][e] = (_Float16)(qp[h] * kScale);
    }
  }

  // ---- online softmax state ------------------------------------------------
  float mRun = -3.0e30f;
  float lRun = 0.0f;
  v8f o[8];
#pragma unroll
  for (int t = 0; t < 8; ++t) {
#pragma unroll
    for (int e = 0; e < 8; ++e) o[t][e] = 0.0f;
  }

  const int rowBase = wave * kRowsPerWave;

  for (int it = 0; it < kRowsPerWave / kTile; ++it) {
    const int tile = rowBase + it * kTile;

    if (it + 1 < kRowsPerWave / kTile) {
      __builtin_prefetch(kp + (size_t)(tile + kTile + l16) * kH + lane, 0, 0);
      __builtin_prefetch(vp + (size_t)(tile + kTile + l16) * kH + lane, 0, 0);
    }

    // ---- scores for rows [tile, tile+32): two 16-row WMMA chains ----------
    v8f cs0, cs1;
#pragma unroll
    for (int e = 0; e < 8; ++e) { cs0[e] = 0.0f; cs1[e] = 0.0f; }

#pragma unroll
    for (int c = 0; c < 4; ++c) {
      // B 32x16, same contiguous k order as A: lane covers col n=l16, K-dim = half*16 + e
      // -> each lane reads 16 contiguous floats of one K row.
      const float* kr0 = kp + (size_t)(tile + l16) * kH + c * 32 + half * 16;
      const float* kr1 = kr0 + (size_t)16 * kH;
      v16h bk0, bk1;
#pragma unroll
      for (int x = 0; x < 4; ++x) {
        const float4 f0 = reinterpret_cast<const float4*>(kr0)[x];
        const float4 f1 = reinterpret_cast<const float4*>(kr1)[x];
        bk0[4 * x + 0] = (_Float16)f0.x; bk0[4 * x + 1] = (_Float16)f0.y;
        bk0[4 * x + 2] = (_Float16)f0.z; bk0[4 * x + 3] = (_Float16)f0.w;
        bk1[4 * x + 0] = (_Float16)f1.x; bk1[4 * x + 1] = (_Float16)f1.y;
        bk1[4 * x + 2] = (_Float16)f1.z; bk1[4 * x + 3] = (_Float16)f1.w;
      }
      cs0 = __builtin_amdgcn_wmma_f32_16x16x32_f16(false, aq[c], false, bk0,
                                                   (short)0, cs0, false, false);
      cs1 = __builtin_amdgcn_wmma_f32_16x16x32_f16(false, aq[c], false, bk1,
                                                   (short)0, cs1, false, false);
    }

    // lane l holds score of row (tile + l%16) / (tile+16 + l%16), replicated.
    float s0 = cs0[0] + mp[tile + l16];
    float s1 = cs1[0] + mp[tile + 16 + l16];

    // tile max (wave32 xor reduction; halves are duplicates so this is exact)
    float tm = fmaxf(s0, s1);
#pragma unroll
    for (int off = 16; off >= 1; off >>= 1)
      tm = fmaxf(tm, __shfl_xor(tm, off, 32));

    const float mNew  = fmaxf(mRun, tm);
    const float alpha = __expf(mRun - mNew);
    const float p0    = __expf(s0 - mNew);
    const float p1    = __expf(s1 - mNew);

    float ps = p0 + p1;
#pragma unroll
    for (int off = 16; off >= 1; off >>= 1)
      ps += __shfl_xor(ps, off, 32);
    lRun = lRun * alpha + 0.5f * ps;   // each row counted twice across halves
    mRun = mNew;

#pragma unroll
    for (int t = 0; t < 8; ++t) {
#pragma unroll
      for (int e = 0; e < 8; ++e) o[t][e] *= alpha;
    }

    // ---- A fragment of probabilities: A[m][kk] = p[kk], all rows equal ----
    v16h ap;
#pragma unroll
    for (int e = 0; e < 16; ++e) {
      const int   src = (e & 7) + half * 8;        // source lane for p value
      const float val = (e < 8) ? p0 : p1;         // uniform select per e
      ap[e] = (_Float16)__shfl(val, src, 32);
    }

    // ---- o[ht] += P x V (8 h-tiles of 16 columns) -------------------------
#pragma unroll
    for (int ht = 0; ht < 8; ++ht) {
      const float* vr = vp + (size_t)(tile + half * 8) * kH + ht * 16 + l16;
      v16h bv;
#pragma unroll
      for (int e = 0; e < 16; ++e)
        bv[e] = (_Float16)vr[(size_t)(e + (e & 8)) * kH];
      o[ht] = __builtin_amdgcn_wmma_f32_16x16x32_f16(false, ap, false, bv,
                                                     (short)0, o[ht], false, false);
    }
  }

  // ---- cross-wave merge through LDS ---------------------------------------
  __shared__ float smO[kWaves][kH];
  __shared__ float smM[kWaves];
  __shared__ float smL[kWaves];

  if (half == 0) {
#pragma unroll
    for (int t = 0; t < 8; ++t) smO[wave][t * 16 + l16] = o[t][0];
    if (l16 == 0) { smM[wave] = mRun; smL[wave] = lRun; }
  }
  __syncthreads();

  if (tid < kH) {
    float gm = smM[0];
#pragma unroll
    for (int w = 1; w < kWaves; ++w) gm = fmaxf(gm, smM[w]);
    float accL = 0.0f, accO = 0.0f;
#pragma unroll
    for (int w = 0; w < kWaves; ++w) {
      const float wgt = __expf(smM[w] - gm);
      accL += wgt * smL[w];
      accO += wgt * smO[w][tid];
    }
    og[(size_t)bn * kH + tid] = accO / accL;
  }
}

extern "C" void kernel_launch(void* const* d_in, const int* in_sizes, int n_in,
                              void* d_out, int out_size, void* d_ws, size_t ws_size,
                              hipStream_t stream) {
  const float* q = (const float*)d_in[0];
  const float* k = (const float*)d_in[1];
  const float* v = (const float*)d_in[2];
  const float* m = (const float*)d_in[3];
  float* out = (float*)d_out;

  fa_decode_wmma<<<dim3(kB * kN), dim3(kWaves * 32), 0, stream>>>(q, k, v, m, out);
}
